// DGCNNTransformer_23184233463984
// MI455X (gfx1250) — matrix operand order repaired and checked by reading, not compiled, not run on hardware
//
#include <hip/hip_runtime.h>
#include <hip/hip_bf16.h>

// ---------------------------------------------------------------------------
// DGCNN + SWA + Transformer forward for MI455X (gfx1250, wave32, WMMA).
// Dense einsums route through one LDS-staged wave32 WMMA GEMM kernel:
//   global_load_b128 (f32) -> v_cvt_pk_f16_f32 -> ds_store -> ds_load_b128
//   -> v_wmma_f32_16x16x32_f16 (x2 per wave per K-step).
// Edge-convs are factored through the gather so GEMMs avoid the K=20 neighbor
// blowup; attention is chunked per batch (16 MB score buffer, L2-resident).
// ---------------------------------------------------------------------------

typedef __attribute__((ext_vector_type(16))) _Float16 v16h;
typedef __attribute__((ext_vector_type(8)))  _Float16 v8h;
typedef __attribute__((ext_vector_type(8)))  float    v8f;

constexpr int   KNN_K    = 20;
constexpr float BN_INV_C = 0.9999950000374997f;  // 1/sqrt(1+1e-5)
constexpr float LN_EPS   = 1e-5f;

// ---------------------------------------------------------------------------
// WMMA GEMM: Out[m,n] = epi( alpha * sum_k A[m,k] * B(k,n) )
//   transB=1: B(k,n) = W[n*ldb + k]   (weights stored [N,K] row-major)
//   transB=0: B(k,n) = W[k*ldb + n]   (plain [K,N], e.g. attn*V)
// Batched via blockIdx.z: b1 = z/nb2, b2 = z%nb2, independent strides.
// Block = 128 threads (4 waves). Block tile = 64(M) x 32(N); each wave owns a
// 16x32 strip (two 16x16 WMMA accumulators). K stepped by 32 through LDS
// panels in fp16. Requires N%32==0, K%32==0 (true for every call site);
// M handled with a per-wave guard (uniform branch keeps EXEC all-ones).
// Epilogue: x*=alpha; if bias x+=bias[n]; if gamma x=gamma[n]*x*BN_INV+beta[n];
//           act: 0 none, 1 relu, 2 leaky-relu(0.2).
// ---------------------------------------------------------------------------
__global__ void __launch_bounds__(128) gemm_wmma_kernel(
    const float* __restrict__ A, int lda, long long sA1, long long sA2,
    const float* __restrict__ Bm, int ldb, long long sB1, long long sB2,
    float* __restrict__ Out, int ldo, long long sO1, long long sO2,
    int M, int Kdim, int nb2, int transB,
    const float* __restrict__ bias, const float* __restrict__ gamma,
    const float* __restrict__ beta, float alpha, int act)
{
    __shared__ _Float16 lA[64 * 32];   // [m within block][k within step]
    __shared__ _Float16 lB[32 * 32];   // [k within step][n within block]

    const int t    = threadIdx.x;      // 0..127
    const int lane = t & 31;
    const int wv   = t >> 5;           // wave 0..3
    const int m0b  = blockIdx.y * 64;
    const int n0b  = blockIdx.x * 32;
    const int b1 = blockIdx.z / nb2;
    const int b2 = blockIdx.z % nb2;

    const float* Ab = A  + b1 * sA1 + b2 * sA2;
    const float* Bb = Bm + b1 * sB1 + b2 * sB2;
    float*       Ob = Out + b1 * sO1 + b2 * sO2;

    const int  m0     = m0b + wv * 16;
    const bool active = (m0 < M);
    const int  r      = lane & 15;                  // A row in tile / D column
    const int  kof    = (lane < 16) ? 0 : 8;        // A-fragment K sub-offset

    v8f acc0 = {0.f, 0.f, 0.f, 0.f, 0.f, 0.f, 0.f, 0.f};
    v8f acc1 = {0.f, 0.f, 0.f, 0.f, 0.f, 0.f, 0.f, 0.f};

    for (int kt = 0; kt < Kdim; kt += 32) {
        // ---- stage A panel (64 rows x 32 k) : coalesced b128, pk-cvt to f16
        for (int i = t; i < 512; i += 128) {
            int row = i >> 3, c4 = (i & 7) << 2;
            int gm = m0b + row;
            float4 v = make_float4(0.f, 0.f, 0.f, 0.f);
            if (gm < M) {
                const float* src = Ab + (long long)gm * lda + kt + c4;
                v = *(const float4*)src;
                __builtin_prefetch(src + 32, 0, 3);      // next K panel
            }
            _Float16* d = &lA[row * 32 + c4];
            d[0] = (_Float16)v.x; d[1] = (_Float16)v.y;
            d[2] = (_Float16)v.z; d[3] = (_Float16)v.w;
        }
        // ---- stage B panel (32 k x 32 n)
        if (transB) {
            for (int i = t; i < 256; i += 128) {
                int n = i >> 3, k4 = (i & 7) << 2;
                float4 v = *(const float4*)(Bb + (long long)(n0b + n) * ldb + kt + k4);
                lB[(k4 + 0) * 32 + n] = (_Float16)v.x;
                lB[(k4 + 1) * 32 + n] = (_Float16)v.y;
                lB[(k4 + 2) * 32 + n] = (_Float16)v.z;
                lB[(k4 + 3) * 32 + n] = (_Float16)v.w;
            }
        } else {
            for (int i = t; i < 256; i += 128) {
                int k = i >> 3, n4 = (i & 7) << 2;
                float4 v = *(const float4*)(Bb + (long long)(kt + k) * ldb + n0b + n4);
                _Float16* d = &lB[k * 32 + n4];
                d[0] = (_Float16)v.x; d[1] = (_Float16)v.y;
                d[2] = (_Float16)v.z; d[3] = (_Float16)v.w;
            }
        }
        __syncthreads();

        if (active) {
            // A fragment: two contiguous 8xf16 runs per lane (ISA 7.12.2)
            const _Float16* ar = &lA[(wv * 16 + r) * 32];
            v8h a0 = *(const v8h*)(ar + kof);
            v8h a1 = *(const v8h*)(ar + kof + 16);
            v16h af;
#pragma unroll
            for (int e = 0; e < 8; ++e) { af[e] = a0[e]; af[8 + e] = a1[e]; }

            // B fragments: lane = N column; K in same order as A fragment
            const _Float16* bc = &lB[r];
            v16h bf0, bf1;
#pragma unroll
            for (int e = 0; e < 8; ++e) {
                bf0[e] = bc[(kof + e) * 32];      bf0[8 + e] = bc[(kof + 16 + e) * 32];
                bf1[e] = bc[(kof + e) * 32 + 16]; bf1[8 + e] = bc[(kof + 16 + e) * 32 + 16];
            }
            acc0 = __builtin_amdgcn_wmma_f32_16x16x32_f16(
                false, af, false, bf0, (short)0, acc0, false, false);
            acc1 = __builtin_amdgcn_wmma_f32_16x16x32_f16(
                false, af, false, bf1, (short)0, acc1, false, false);
        }
        __syncthreads();
    }

    if (!active) return;
    const int mbase = m0 + ((lane < 16) ? 0 : 8);   // C/D: VGPR v -> M = v (+8)
    const int nlo = n0b + r, nhi = n0b + 16 + r;
#pragma unroll
    for (int v = 0; v < 8; ++v) {
        float x0 = acc0[v] * alpha;
        float x1 = acc1[v] * alpha;
        if (bias)  { x0 += bias[nlo];  x1 += bias[nhi]; }
        if (gamma) {
            x0 = gamma[nlo] * x0 * BN_INV_C + beta[nlo];
            x1 = gamma[nhi] * x1 * BN_INV_C + beta[nhi];
        }
        if (act == 1)      { x0 = fmaxf(x0, 0.f); x1 = fmaxf(x1, 0.f); }
        else if (act == 2) { x0 = (x0 > 0.f) ? x0 : 0.2f * x0;
                             x1 = (x1 > 0.f) ? x1 : 0.2f * x1; }
        long long rowoff = (long long)(mbase + v) * ldo;
        Ob[rowoff + nlo] = x0;
        Ob[rowoff + nhi] = x1;
    }
}

// ---------------------------------------------------------------------------
// [B,3,N] -> [B,N,3]
// ---------------------------------------------------------------------------
__global__ void transpose_kernel(const float* __restrict__ x,
                                 float* __restrict__ pts, int N)
{
    int tid = blockIdx.x * blockDim.x + threadIdx.x;   // over B*N
    int b = tid / N, n = tid % N;
    const float* xb = x + (long long)b * 3 * N;
    float* p = pts + (long long)tid * 3;
    p[0] = xb[n];
    p[1] = xb[N + n];
    p[2] = xb[2 * N + n];
}

// ---------------------------------------------------------------------------
// kNN: per-thread query, LDS-tiled candidates, sorted top-20 of -||q-m||^2.
// Only the neighbor *set* matters downstream (all consumers max over k).
// ---------------------------------------------------------------------------
template <int C>
__global__ void knn_kernel(const float* __restrict__ X, int* __restrict__ out, int N)
{
    constexpr int BT = 128, TM = 32;
    const int b = blockIdx.y;
    const int n = blockIdx.x * BT + threadIdx.x;
    const float* Xb = X + (long long)b * N * C;

    float q[C];
#pragma unroll
    for (int c = 0; c < C; ++c) q[c] = Xb[(long long)n * C + c];

    float bestd[KNN_K];
    int   besti[KNN_K];
#pragma unroll
    for (int i = 0; i < KNN_K; ++i) { bestd[i] = -3.4e38f; besti[i] = 0; }

    __shared__ float tile[TM * C];
    for (int mt = 0; mt < N; mt += TM) {
        __syncthreads();
        for (int i = threadIdx.x; i < TM * C; i += BT)
            tile[i] = Xb[(long long)mt * C + i];
        __syncthreads();
        for (int j = 0; j < TM; ++j) {
            float d = 0.f;
#pragma unroll
            for (int c = 0; c < C; ++c) {
                float t = q[c] - tile[j * C + c];
                d += t * t;
            }
            d = -d;                           // reference: topk of -sqdist
            if (d > bestd[KNN_K - 1]) {
                int pos = KNN_K - 1;
                while (pos > 0 && bestd[pos - 1] < d) {
                    bestd[pos] = bestd[pos - 1];
                    besti[pos] = besti[pos - 1];
                    --pos;
                }
                bestd[pos] = d;
                besti[pos] = mt + j;
            }
        }
    }
    int* o = out + ((long long)b * N + n) * KNN_K;
#pragma unroll
    for (int i = 0; i < KNN_K; ++i) o[i] = besti[i];
}

// ---------------------------------------------------------------------------
// conv1 partial products (C=3, below the WMMA K=32 granule):
// y1 = p . w[:, :3]^T, y2 = p . w[:, 3:6]^T
// ---------------------------------------------------------------------------
__global__ void conv1_y_kernel(const float* __restrict__ pts,
                               const float* __restrict__ w,
                               float* __restrict__ y1, float* __restrict__ y2)
{
    long long bn = blockIdx.x;
    int o = threadIdx.x;                     // 64 channels
    float p0 = pts[bn * 3], p1 = pts[bn * 3 + 1], p2 = pts[bn * 3 + 2];
    const float* wr = w + o * 6;
    y1[bn * 64 + o] = p0 * wr[0] + p1 * wr[1] + p2 * wr[2];
    y2[bn * 64 + o] = p0 * wr[3] + p1 * wr[4] + p2 * wr[5];
}

// ---------------------------------------------------------------------------
// Edge-conv epilogue: out = max_k lrelu(bn(y1[idx[k]] - y1[n] + y2[n]))
// ---------------------------------------------------------------------------
__global__ void edge_max_kernel(const float* __restrict__ y1,
                                const float* __restrict__ y2,
                                const int* __restrict__ idx,
                                const float* __restrict__ g,
                                const float* __restrict__ bb,
                                float* __restrict__ out, int N, int O)
{
    long long bn = blockIdx.x;
    int b = (int)(bn / N);
    int o = threadIdx.x;
    __shared__ int sidx[KNN_K];
    if (threadIdx.x < KNN_K) sidx[threadIdx.x] = idx[bn * KNN_K + threadIdx.x];
    __syncthreads();

    const float* y1b = y1 + (long long)b * N * O;
    float c1 = y1[bn * O + o];
    float c2 = y2[bn * O + o];
    float sc = g[o] * BN_INV_C, be = bb[o];
    float best = -3.4e38f;
#pragma unroll 4
    for (int k = 0; k < KNN_K; ++k) {
        float h = (y1b[(long long)sidx[k] * O + o] - c1 + c2) * sc + be;
        h = (h > 0.f) ? h : 0.2f * h;
        best = fmaxf(best, h);
    }
    out[bn * O + o] = best;
}

// ---------------------------------------------------------------------------
// SWA position encoder stage 1 (6 -> 64 with relu), per batch chunk b.
// ---------------------------------------------------------------------------
__global__ void swa_pe1_kernel(const float* __restrict__ pts,
                               const int* __restrict__ idx,
                               const float* __restrict__ pw1,
                               const float* __restrict__ pb1,
                               float* __restrict__ pe1, int b, int N)
{
    int n = blockIdx.x, o = threadIdx.x;     // 64 channels
    const float* pb_ = pts + (long long)b * N * 3;
    float c0 = pb_[n * 3], c1 = pb_[n * 3 + 1], c2 = pb_[n * 3 + 2];
    const float* wr = pw1 + o * 6;
    float w0 = wr[0], w1 = wr[1], w2 = wr[2], w3 = wr[3], w4 = wr[4], w5 = wr[5];
    float cent = c0 * w3 + c1 * w4 + c2 * w5 + pb1[o];
    const int* ir = idx + ((long long)b * N + n) * KNN_K;
#pragma unroll 4
    for (int k = 0; k < KNN_K; ++k) {
        int m = ir[k];
        float d0 = pb_[m * 3] - c0, d1 = pb_[m * 3 + 1] - c1, d2 = pb_[m * 3 + 2] - c2;
        float v = d0 * w0 + d1 * w1 + d2 * w2 + cent;
        pe1[((long long)n * KNN_K + k) * 64 + o] = fmaxf(v, 0.f);
    }
}

// h1 = relu(bn(z[idx]-z[n] + pemix)) ; per batch chunk b
__global__ void swa_combine_kernel(const float* __restrict__ z,
                                   const float* __restrict__ pemix,
                                   const int* __restrict__ idx,
                                   const float* __restrict__ g,
                                   const float* __restrict__ bb,
                                   float* __restrict__ h1, int b, int N)
{
    int n = blockIdx.x, o = threadIdx.x;
    const float* zb = z + (long long)b * N * 64;
    float zc = zb[(long long)n * 64 + o];
    float sc = g[o] * BN_INV_C, be = bb[o];
    const int* ir = idx + ((long long)b * N + n) * KNN_K;
#pragma unroll 4
    for (int k = 0; k < KNN_K; ++k) {
        float v = (zb[(long long)ir[k] * 64 + o] - zc +
                   pemix[((long long)n * KNN_K + k) * 64 + o]) * sc + be;
        h1[((long long)n * KNN_K + k) * 64 + o] = fmaxf(v, 0.f);
    }
}

// out[b,n,:] = max_k h2[n,k,:] + x[b,n,:]
__global__ void swa_maxadd_kernel(const float* __restrict__ h2,
                                  const float* __restrict__ x,
                                  float* __restrict__ out, int b, int N)
{
    int n = blockIdx.x, o = threadIdx.x;
    float best = -3.4e38f;
#pragma unroll 4
    for (int k = 0; k < KNN_K; ++k)
        best = fmaxf(best, h2[((long long)n * KNN_K + k) * 64 + o]);
    long long off = ((long long)b * N + n) * 64 + o;
    out[off] = best + x[off];
}

// concat [x1|x2|x3|x4] -> [B*N, 512]
__global__ void concat_kernel(const float* __restrict__ x1, const float* __restrict__ x2,
                              const float* __restrict__ x3, const float* __restrict__ x4,
                              float* __restrict__ out)
{
    long long bn = blockIdx.x;
    int t = threadIdx.x;                     // 512
    float v;
    if (t < 64)        v = x1[bn * 64 + t];
    else if (t < 128)  v = x2[bn * 64 + (t - 64)];
    else if (t < 256)  v = x3[bn * 128 + (t - 128)];
    else               v = x4[bn * 256 + (t - 256)];
    out[bn * 512 + t] = v;
}

// row softmax, L = 1024, block 256
__global__ void softmax_kernel(float* __restrict__ sc, int L)
{
    long long row = blockIdx.x;
    float* p = sc + row * L;
    int t = threadIdx.x;
    __shared__ float red[256];

    float mx = -3.4e38f;
    for (int i = t; i < L; i += 256) mx = fmaxf(mx, p[i]);
    red[t] = mx; __syncthreads();
    for (int s = 128; s > 0; s >>= 1) { if (t < s) red[t] = fmaxf(red[t], red[t + s]); __syncthreads(); }
    mx = red[0]; __syncthreads();

    float sum = 0.f;
    for (int i = t; i < L; i += 256) { float e = __expf(p[i] - mx); p[i] = e; sum += e; }
    red[t] = sum; __syncthreads();
    for (int s = 128; s > 0; s >>= 1) { if (t < s) red[t] += red[t + s]; __syncthreads(); }
    float inv = 1.f / red[0];
    __syncthreads();
    for (int i = t; i < L; i += 256) p[i] *= inv;
}

// out = LN(x + y) with gamma/beta ; block = D = 256 (one thread per channel)
__global__ void add_ln_kernel(const float* __restrict__ x, const float* __restrict__ y,
                              const float* __restrict__ g, const float* __restrict__ b_,
                              float* __restrict__ out, int D)
{
    long long row = blockIdx.x;
    int t = threadIdx.x;
    float v = x[row * D + t] + y[row * D + t];
    __shared__ float red[256];
    red[t] = v; __syncthreads();
    for (int s = 128; s > 0; s >>= 1) { if (t < s) red[t] += red[t + s]; __syncthreads(); }
    float mean = red[0] / D; __syncthreads();
    float d = v - mean;
    red[t] = d * d; __syncthreads();
    for (int s = 128; s > 0; s >>= 1) { if (t < s) red[t] += red[t + s]; __syncthreads(); }
    float var = red[0] / D;
    out[row * D + t] = d * rsqrtf(var + LN_EPS) * g[t] + b_[t];
}

// global max + mean pool over N ; grid (C/256, B)
__global__ void pool_kernel(const float* __restrict__ h, float* __restrict__ out,
                            int N, int C)
{
    int b = blockIdx.y;
    int c = blockIdx.x * blockDim.x + threadIdx.x;
    const float* p = h + (long long)b * N * C + c;
    float mx = -3.4e38f, sum = 0.f;
    for (int n = 0; n < N; ++n) { float v = p[(long long)n * C]; mx = fmaxf(mx, v); sum += v; }
    out[(long long)b * 2 * C + c]     = mx;
    out[(long long)b * 2 * C + C + c] = sum / (float)N;
}

// final FC (N=40 not a multiple of 16 -> plain VALU) : out = A.W^T + bias
__global__ void fc_small_kernel(const float* __restrict__ A, const float* __restrict__ W,
                                const float* __restrict__ bias, float* __restrict__ out,
                                int Nn, int Kd)
{
    int m = blockIdx.x;
    int n = threadIdx.x;
    if (n >= Nn) return;
    float s = bias[n];
    for (int k = 0; k < Kd; ++k) s += A[(long long)m * Kd + k] * W[(long long)n * Kd + k];
    out[(long long)m * Nn + n] = s;
}

// ---------------------------------------------------------------------------
// Host side
// ---------------------------------------------------------------------------
static inline void gemm_full(hipStream_t st,
                             const float* A, int lda, long long sA1, long long sA2,
                             const float* W, int ldb, long long sB1, long long sB2,
                             float* O, int ldo, long long sO1, long long sO2,
                             int M, int Nn, int Kd, int nz, int nb2, int transB,
                             const float* bias, const float* gamma, const float* beta,
                             float alpha, int act)
{
    dim3 grid(Nn / 32, (M + 63) / 64, nz);
    gemm_wmma_kernel<<<grid, 128, 0, st>>>(A, lda, sA1, sA2, W, ldb, sB1, sB2,
                                           O, ldo, sO1, sO2, M, Kd, nb2, transB,
                                           bias, gamma, beta, alpha, act);
}

static inline void gemm2(hipStream_t st, const float* A, int lda,
                         const float* W, int ldw, float* O, int ldo,
                         int M, int Nn, int Kd,
                         const float* bias, const float* gamma, const float* beta, int act)
{
    gemm_full(st, A, lda, 0, 0, W, ldw, 0, 0, O, ldo, 0, 0,
              M, Nn, Kd, 1, 1, 1, bias, gamma, beta, 1.f, act);
}

struct SwaP {
    const float *pw1, *pb1, *pw2, *pb2, *mw1, *mg1, *mb1, *mw2, *mg2, *mb2;
};

extern "C" void kernel_launch(void* const* d_in, const int* in_sizes, int n_in,
                              void* d_out, int out_size, void* d_ws, size_t ws_size,
                              hipStream_t stream)
{
    (void)in_sizes; (void)n_in; (void)out_size; (void)ws_size;
    const int B = 16, N = 1024, H = 4;
    const long long BN = (long long)B * N;

    const float* P[64];
    for (int i = 0; i < 64; ++i) P[i] = (const float*)d_in[i];

    const float* x      = P[0];
    const float* c1_w = P[1];  const float* c1_g = P[2];  const float* c1_b = P[3];
    SwaP swa1 = {P[4], P[5], P[6], P[7], P[8], P[9], P[10], P[11], P[12], P[13]};
    const float* c2_w = P[14]; const float* c2_g = P[15]; const float* c2_b = P[16];
    SwaP swa2 = {P[17], P[18], P[19], P[20], P[21], P[22], P[23], P[24], P[25], P[26]};
    const float* c3_w = P[27]; const float* c3_g = P[28]; const float* c3_b = P[29];
    const float* c4_w = P[30]; const float* c4_g = P[31]; const float* c4_b = P[32];
    const float* fuse_w = P[33]; const float* fuse_g = P[34]; const float* fuse_b = P[35];
    const float* wq = P[36]; const float* bq = P[37];
    const float* wk = P[38]; const float* bk = P[39];
    const float* wv = P[40]; const float* bv = P[41];
    const float* wo = P[42]; const float* bo = P[43];
    const float* ln1_g = P[44]; const float* ln1_b = P[45];
    const float* ln2_g = P[46]; const float* ln2_b = P[47];
    const float* fw1 = P[48]; const float* fb1 = P[49];
    const float* fw2 = P[50]; const float* fb2 = P[51];
    const float* c5_w = P[52]; const float* c5_g = P[53]; const float* c5_b = P[54];
    const float* l1_w = P[55]; const float* bn6_g = P[56]; const float* bn6_b = P[57];
    const float* l2_w = P[58]; const float* l2_b = P[59];
    const float* bn7_g = P[60]; const float* bn7_b = P[61];
    const float* l3_w = P[62]; const float* l3_b = P[63];

    // ---- workspace carve-out (lifetime-based reuse noted inline) ----
    char* base = (char*)d_ws;
    size_t off = 0;
    auto alloc = [&](size_t nElem) -> void* {
        void* p = base + off;
        off += (((size_t)nElem * 4 + 255) & ~(size_t)255);
        return p;
    };
    float* PTS  = (float*)alloc(BN * 3);
    int*   IDXP = (int*)alloc(BN * KNN_K);
    int*   IDX1 = (int*)alloc(BN * KNN_K);
    int*   IDX2 = (int*)alloc(BN * KNN_K);
    int*   IDX3 = (int*)alloc(BN * KNN_K);
    float* XE   = (float*)alloc(BN * 64);        // edge-conv outputs (pre-SWA)
    float* X1   = (float*)alloc(BN * 64);
    float* X2   = (float*)alloc(BN * 64);
    float* X3   = (float*)alloc(BN * 128);
    float* X4   = (float*)alloc(BN * 256);       // later reused as T1 (post-LN1)
    float* Y1   = (float*)alloc(BN * 256);       // later reused as Q
    float* Y2   = (float*)alloc(BN * 256);       // later reused as Kmat
    float* Z    = (float*)alloc(BN * 64);
    float* SWA_A = (float*)alloc((long long)N * KNN_K * 64);   // per-b chunks
    float* SWA_B = (float*)alloc((long long)N * KNN_K * 64);
    float* F    = (float*)alloc(BN * 256);       // fuse out; reused as T2
    float* V    = (float*)alloc(BN * 256);
    float* SC   = (float*)alloc((long long)H * N * N);  // per-b scores; reused as OP
    float* AO   = (float*)alloc(BN * 256);
    float* FF1  = (float*)alloc(BN * 512);       // also holds the 512-wide concat
    float* H5   = (float*)alloc(BN * 1024);
    float* POOL = (float*)alloc((long long)B * 2048);
    float* G1   = (float*)alloc((long long)B * 512);
    float* G2   = (float*)alloc((long long)B * 256);
    float* Q = Y1, *KM = Y2, *OP = SC, *T1 = X4, *T2 = F;

    // ---- 0) points ----
    transpose_kernel<<<(int)(BN / 256), 256, 0, stream>>>(x, PTS, N);
    knn_kernel<3><<<dim3(N / 128, B), 128, 0, stream>>>(PTS, IDXP, N);

    // ---- edge-conv runner (C in {64,128}) ----
    auto run_edge = [&](const float* xin, int C, const float* w, const float* g,
                        const float* bb, int O, int* idx, float* xout) {
        if (C == 64) knn_kernel<64><<<dim3(N / 128, B), 128, 0, stream>>>(xin, idx, N);
        else         knn_kernel<128><<<dim3(N / 128, B), 128, 0, stream>>>(xin, idx, N);
        gemm2(stream, xin, C, w,     2 * C, Y1, O, (int)BN, O, C, nullptr, nullptr, nullptr, 0);
        gemm2(stream, xin, C, w + C, 2 * C, Y2, O, (int)BN, O, C, nullptr, nullptr, nullptr, 0);
        edge_max_kernel<<<(int)BN, O, 0, stream>>>(Y1, Y2, idx, g, bb, xout, N, O);
    };

    // ---- SWA runner (per-batch chunked to keep [N*K,64] buffers small) ----
    auto run_swa = [&](const float* xin, const SwaP& s, float* xout) {
        // z = x . mw1[:, :64]^T   (full batch, feeds the (z[idx]-z) gather)
        gemm2(stream, xin, 64, s.mw1, 128, Z, 64, (int)BN, 64, 64, nullptr, nullptr, nullptr, 0);
        for (int b = 0; b < B; ++b) {
            swa_pe1_kernel<<<N, 64, 0, stream>>>(PTS, IDXP, s.pw1, s.pb1, SWA_A, b, N);
            gemm2(stream, SWA_A, 64, s.pw2, 64, SWA_B, 64, N * KNN_K, 64, 64,
                  s.pb2, nullptr, nullptr, 0);                      // pe2
            gemm2(stream, SWA_B, 64, s.mw1 + 64, 128, SWA_A, 64, N * KNN_K, 64, 64,
                  nullptr, nullptr, nullptr, 0);                    // pe . mw1[:,64:]^T
            swa_combine_kernel<<<N, 64, 0, stream>>>(Z, SWA_A, IDXP, s.mg1, s.mb1,
                                                     SWA_B, b, N);  // h1
            gemm2(stream, SWA_B, 64, s.mw2, 64, SWA_A, 64, N * KNN_K, 64, 64,
                  nullptr, s.mg2, s.mb2, 1);                        // h2 = relu(bn(.))
            swa_maxadd_kernel<<<N, 64, 0, stream>>>(SWA_A, xin, xout, b, N);
        }
    };

    // ---- 1) edge_conv1 (C=3 via VALU partials) + swa1 ----
    conv1_y_kernel<<<(int)BN, 64, 0, stream>>>(PTS, c1_w, Y1, Y2);
    edge_max_kernel<<<(int)BN, 64, 0, stream>>>(Y1, Y2, IDXP, c1_g, c1_b, XE, N, 64);
    run_swa(XE, swa1, X1);

    // ---- 2) edge_conv2 + swa2 ----
    run_edge(X1, 64, c2_w, c2_g, c2_b, 64, IDX1, XE);
    run_swa(XE, swa2, X2);

    // ---- 3) edge_conv3 / edge_conv4 ----
    run_edge(X2, 64,  c3_w, c3_g, c3_b, 128, IDX2, X3);
    run_edge(X3, 128, c4_w, c4_g, c4_b, 256, IDX3, X4);

    // ---- 4) multi-scale fusion: concat(512) -> 256, BN + lrelu ----
    concat_kernel<<<(int)BN, 512, 0, stream>>>(X1, X2, X3, X4, FF1);
    gemm2(stream, FF1, 512, fuse_w, 512, F, 256, (int)BN, 256, 512,
          nullptr, fuse_g, fuse_b, 2);

    // ---- 5) transformer ----
    gemm2(stream, F, 256, wq, 256, Q, 256, (int)BN, 256, 256, bq, nullptr, nullptr, 0);
    gemm2(stream, F, 256, wk, 256, KM, 256, (int)BN, 256, 256, bk, nullptr, nullptr, 0);
    gemm2(stream, F, 256, wv, 256, V, 256, (int)BN, 256, 256, bv, nullptr, nullptr, 0);
    for (int b = 0; b < B; ++b) {
        const float* Qb = Q  + (long long)b * N * 256;
        const float* Kb = KM + (long long)b * N * 256;
        const float* Vb = V  + (long long)b * N * 256;
        // scores[h,n,m] = 0.125 * sum_d q[n,h*64+d] k[m,h*64+d]
        gemm_full(stream, Qb, 256, 0, 64, Kb, 256, 0, 64,
                  SC, N, 0, (long long)N * N,
                  N, N, 64, H, H, 1, nullptr, nullptr, nullptr, 0.125f, 0);
        softmax_kernel<<<H * N, 256, 0, stream>>>(SC, N);
        // out[n,h*64+d] = sum_m a[h,n,m] v[m,h*64+d]   (transB=0)
        gemm_full(stream, SC, N, 0, (long long)N * N, Vb, 256, 0, 64,
                  AO + (long long)b * N * 256, 256, 0, 64,
                  N, 64, N, H, H, 0, nullptr, nullptr, nullptr, 1.f, 0);
    }
    gemm2(stream, AO, 256, wo, 256, OP, 256, (int)BN, 256, 256, bo, nullptr, nullptr, 0);
    add_ln_kernel<<<(int)BN, 256, 0, stream>>>(F, OP, ln1_g, ln1_b, T1, 256);
    gemm2(stream, T1, 256, fw1, 256, FF1, 512, (int)BN, 512, 256, fb1, nullptr, nullptr, 1);
    gemm2(stream, FF1, 512, fw2, 512, OP, 256, (int)BN, 256, 512, fb2, nullptr, nullptr, 0);
    add_ln_kernel<<<(int)BN, 256, 0, stream>>>(T1, OP, ln2_g, ln2_b, T2, 256);

    // ---- 6) head: conv5 -> max/mean pool -> FCs ----
    gemm2(stream, T2, 256, c5_w, 256, H5, 1024, (int)BN, 1024, 256,
          nullptr, c5_g, c5_b, 2);
    pool_kernel<<<dim3(1024 / 256, B), 256, 0, stream>>>(H5, POOL, N, 1024);
    gemm2(stream, POOL, 2048, l1_w, 2048, G1, 512, B, 512, 2048,
          nullptr, bn6_g, bn6_b, 2);
    gemm2(stream, G1, 512, l2_w, 512, G2, 256, B, 256, 512,
          l2_b, bn7_g, bn7_b, 2);
    fc_small_kernel<<<B, 64, 0, stream>>>(G2, l3_w, l3_b, (float*)d_out, 40, 256);
}
